// Head_76450417869200
// MI455X (gfx1250) — matrix operand order repaired and checked by reading, not compiled, not run on hardware
//
#include <hip/hip_runtime.h>
#include <hip/hip_bf16.h>
#include <math.h>

typedef __attribute__((ext_vector_type(16))) _Float16 v16h;
typedef __attribute__((ext_vector_type(8)))  _Float16 v8h;
typedef __attribute__((ext_vector_type(8)))  float    v8f;

#define N_EMBED 1024
#define HEAD    64
#define TLEN    2048
#define BATCH   8

static __device__ __forceinline__ v8f wmma_f16(v16h a, v16h b, v8f c) {
    return __builtin_amdgcn_wmma_f32_16x16x32_f16(false, a, false, b, (short)0, c, false, false);
}

// A-operand chunk from f32 row: K-runs [8h,8h+8) at p0 and [16+8h,+8) at p0+16
static __device__ __forceinline__ v16h load_a_f32(const float* p0) {
    const float* p1 = p0 + 16;
    v16h a;
#pragma unroll
    for (int j = 0; j < 8; ++j) {
        a[j]     = (_Float16)p0[j];
        a[j + 8] = (_Float16)p1[j];
    }
    return a;
}

// A-operand from f16 memory: two 8-elem runs at p and p+16 (two b128 loads)
static __device__ __forceinline__ v16h load_a_f16(const _Float16* p) {
    v8h lo = *(const v8h*)p;
    v8h hi = *(const v8h*)(p + 16);
    v16h a;
#pragma unroll
    for (int j = 0; j < 8; ++j) { a[j] = lo[j]; a[j + 8] = hi[j]; }
    return a;
}

// B-operand from f16 memory: two 8-elem runs at p and p+16 (two b128 loads)
static __device__ __forceinline__ v16h load_b_f16(const _Float16* p) {
    v8h lo = *(const v8h*)p;
    v8h hi = *(const v8h*)(p + 16);
    v16h b;
#pragma unroll
    for (int j = 0; j < 8; ++j) { b[j] = lo[j]; b[j + 8] = hi[j]; }
    return b;
}

// ---------------------------------------------------------------------------
// Kernel 0: Wq/Wk/Wv [E,H] f32  ->  WT [3][H][E] f16 (transposed + converted)
// ---------------------------------------------------------------------------
__global__ void wtrans_kernel(const float* __restrict__ Wq,
                              const float* __restrict__ Wk,
                              const float* __restrict__ Wv,
                              _Float16* __restrict__ WT) {
    int i = blockIdx.x * blockDim.x + threadIdx.x;
    const int per = N_EMBED * HEAD;
    if (i >= 3 * per) return;
    int kind = i / per;
    int r    = i % per;              // h * N_EMBED + e
    int h    = r / N_EMBED;
    int e    = r % N_EMBED;
    const float* W = (kind == 0) ? Wq : (kind == 1) ? Wk : Wv;
    WT[i] = (_Float16)W[e * HEAD + h];
}

// ---------------------------------------------------------------------------
// Kernel 1a: q = index @ Wq -> q16 [B][T][H] f16. One wave per 16x64 tile.
// ---------------------------------------------------------------------------
__global__ __launch_bounds__(32) void proj_q_kernel(
    const float* __restrict__ index, const _Float16* __restrict__ WqT,
    _Float16* __restrict__ q16) {

    const int lane = threadIdx.x;
    const int half = lane >> 4;
    const int lm   = lane & 15;

    const int tilesPerB = TLEN / 16;
    int b    = blockIdx.x / tilesPerB;
    int tile = blockIdx.x % tilesPerB;
    int tbase = tile * 16;

    const float* xrow = index + ((size_t)b * TLEN + tbase + lm) * N_EMBED;

    v8f acc[4]; acc[0] = 0.0f; acc[1] = 0.0f; acc[2] = 0.0f; acc[3] = 0.0f;

    for (int c = 0; c < N_EMBED / 32; ++c) {
        v16h a = load_a_f32(xrow + c * 32 + half * 8);
        int kb0 = c * 32 + half * 8;
#pragma unroll
        for (int nt = 0; nt < 4; ++nt) {
            v16h bm = load_b_f16(WqT + (size_t)(nt * 16 + lm) * N_EMBED + kb0);
            acc[nt] = wmma_f16(a, bm, acc[nt]);
        }
    }

    _Float16* out = q16 + (size_t)b * TLEN * HEAD;
#pragma unroll
    for (int nt = 0; nt < 4; ++nt) {
        int n = nt * 16 + lm;
#pragma unroll
        for (int v = 0; v < 8; ++v)
            out[(size_t)(tbase + v + half * 8) * HEAD + n] = (_Float16)acc[nt][v];
    }
}

// ---------------------------------------------------------------------------
// Kernel 1b: fused k & v projections (memory read once, 8 WMMAs per A-load).
//   k = memory @ Wk -> k16  [B][T][H] row-major f16
//   v = memory @ Wv -> vT16 [B][H][T] transposed f16
// ---------------------------------------------------------------------------
__global__ __launch_bounds__(32) void proj_kv_kernel(
    const float* __restrict__ memory,
    const _Float16* __restrict__ WkT, const _Float16* __restrict__ WvT,
    _Float16* __restrict__ k16, _Float16* __restrict__ vT16) {

    const int lane = threadIdx.x;
    const int half = lane >> 4;
    const int lm   = lane & 15;

    const int tilesPerB = TLEN / 16;
    int b    = blockIdx.x / tilesPerB;
    int tile = blockIdx.x % tilesPerB;
    int tbase = tile * 16;

    const float* xrow = memory + ((size_t)b * TLEN + tbase + lm) * N_EMBED;

    v8f ak[4], av[4];
#pragma unroll
    for (int nt = 0; nt < 4; ++nt) { ak[nt] = 0.0f; av[nt] = 0.0f; }

    for (int c = 0; c < N_EMBED / 32; ++c) {
        v16h a = load_a_f32(xrow + c * 32 + half * 8);
        int kb0 = c * 32 + half * 8;
#pragma unroll
        for (int nt = 0; nt < 4; ++nt) {
            v16h bk = load_b_f16(WkT + (size_t)(nt * 16 + lm) * N_EMBED + kb0);
            ak[nt] = wmma_f16(a, bk, ak[nt]);
        }
#pragma unroll
        for (int nt = 0; nt < 4; ++nt) {
            v16h bv = load_b_f16(WvT + (size_t)(nt * 16 + lm) * N_EMBED + kb0);
            av[nt] = wmma_f16(a, bv, av[nt]);
        }
    }

    // k: row-major f16 (C/D layout: lane -> col n; vgpr v -> row v + 8*half)
    _Float16* kout = k16 + (size_t)b * TLEN * HEAD;
#pragma unroll
    for (int nt = 0; nt < 4; ++nt) {
        int n = nt * 16 + lm;
#pragma unroll
        for (int v = 0; v < 8; ++v)
            kout[(size_t)(tbase + v + half * 8) * HEAD + n] = (_Float16)ak[nt][v];
    }
    // v: transposed [H][T]; lane's 8 rows are consecutive t -> one b128 store
    _Float16* vout = vT16 + (size_t)b * HEAD * TLEN;
#pragma unroll
    for (int nt = 0; nt < 4; ++nt) {
        int n = nt * 16 + lm;
        v8h pk;
#pragma unroll
        for (int v = 0; v < 8; ++v) pk[v] = (_Float16)av[nt][v];
        *(v8h*)(vout + (size_t)n * TLEN + tbase + half * 8) = pk;
    }
}

// ---------------------------------------------------------------------------
// Kernel 2: fused causal flash attention. One wave per (batch, 16-query tile).
// Computes S^T = K*Q^T so the C-layout of S^T coincides with the A-layout of P.
// ---------------------------------------------------------------------------
__global__ __launch_bounds__(32) void attn_kernel(
    const _Float16* __restrict__ q16, const _Float16* __restrict__ k16,
    const _Float16* __restrict__ vT16, float* __restrict__ out) {

    const int lane = threadIdx.x;
    const int half = lane >> 4;
    const int lm   = lane & 15;

    const int tilesPerB = TLEN / 16;
    int b    = blockIdx.x / tilesPerB;
    int tile = blockIdx.x % tilesPerB;
    int qbase = tile * 16;

    const _Float16* qb = q16  + (size_t)b * TLEN * HEAD;
    const _Float16* kb = k16  + (size_t)b * TLEN * HEAD;
    const _Float16* vb = vT16 + (size_t)b * HEAD * TLEN;

    // Q^T as B operand (N=query, K=head). Hoisted: reused by every key block.
    v16h bq[2];
#pragma unroll
    for (int c = 0; c < 2; ++c)
        bq[c] = load_b_f16(qb + (size_t)(qbase + lm) * HEAD + c * 32 + half * 8);

    v8f o[4]; o[0] = 0.0f; o[1] = 0.0f; o[2] = 0.0f; o[3] = 0.0f;
    float m = -INFINITY;
    float l = 0.0f;
    const float scale = 0.03125f;      // 1/sqrt(N_EMBED)
    const int query = qbase + lm;      // this lane's query row

    int jmax = (qbase + 15) / 32;      // last (and only partially-masked) block
    for (int jb = 0; jb <= jmax; ++jb) {
        int kbase = jb * 32;

        // S^T: two 16(keys) x 16(queries) tiles, K-reduce over head dim (2x32)
        v8f st0 = 0.0f, st1 = 0.0f;
#pragma unroll
        for (int c = 0; c < 2; ++c) {
            v16h a0 = load_a_f16(kb + (size_t)(kbase + lm) * HEAD + c * 32 + half * 8);
            v16h a1 = load_a_f16(kb + (size_t)(kbase + 16 + lm) * HEAD + c * 32 + half * 8);
            st0 = wmma_f16(a0, bq[c], st0);
            st1 = wmma_f16(a1, bq[c], st1);
        }

        // Issue V-tile loads NOW so their latency hides under the softmax VALU.
        v16h bv[4];
#pragma unroll
        for (int nt = 0; nt < 4; ++nt)
            bv[nt] = load_b_f16(vb + (size_t)(nt * 16 + lm) * TLEN + kbase + half * 8);

        // Per-lane scores: lane holds query lm, keys kbase + {8h+v, 16+8h+v}
        bool maskblk = (kbase + 31 > qbase);
        float sv[16];
#pragma unroll
        for (int v = 0; v < 8; ++v) {
            float s0 = st0[v] * scale;
            float s1 = st1[v] * scale;
            if (maskblk) {
                if (kbase + half * 8 + v > query)      s0 = -INFINITY;
                if (kbase + 16 + half * 8 + v > query) s1 = -INFINITY;
            }
            sv[v] = s0; sv[8 + v] = s1;
        }

        // Row max (16 local values + partner half-wave)
        float mloc = sv[0];
#pragma unroll
        for (int i = 1; i < 16; ++i) mloc = fmaxf(mloc, sv[i]);
        mloc = fmaxf(mloc, __shfl_xor(mloc, 16));
        float mnew  = fmaxf(m, mloc);
        float alpha = __expf(m - mnew);     // 0 on first block (m=-inf)

        // P = exp(S - m); C-layout of S^T == A-layout of P -> cast in place
        v16h pa;
        float psum = 0.0f;
#pragma unroll
        for (int i = 0; i < 16; ++i) {
            float p = __expf(sv[i] - mnew);
            psum += p;
            pa[i] = (_Float16)p;
        }
        psum += __shfl_xor(psum, 16);
        l = l * alpha + psum;
        m = mnew;

        // Broadcast per-query alpha to O's C-layout rows (row = v + 8*half)
        float asc[8];
#pragma unroll
        for (int v = 0; v < 8; ++v) asc[v] = __shfl(alpha, v + 8 * half);
#pragma unroll
        for (int nt = 0; nt < 4; ++nt)
#pragma unroll
            for (int v = 0; v < 8; ++v) o[nt][v] *= asc[v];

        // O += P * V ; V as B operand (N=head col, K=keys contiguous in vT)
#pragma unroll
        for (int nt = 0; nt < 4; ++nt)
            o[nt] = wmma_f16(pa, bv[nt], o[nt]);
    }

    // Normalize by row sums and store f32 output [B][T][H]
    float linv = 1.0f / l;
    float lsc[8];
#pragma unroll
    for (int v = 0; v < 8; ++v) lsc[v] = __shfl(linv, v + 8 * half);

    float* ob = out + (size_t)b * TLEN * HEAD;
#pragma unroll
    for (int nt = 0; nt < 4; ++nt) {
        int n = nt * 16 + lm;
#pragma unroll
        for (int v = 0; v < 8; ++v) {
            int row = qbase + v + half * 8;
            ob[(size_t)row * HEAD + n] = o[nt][v] * lsc[v];
        }
    }
}

// ---------------------------------------------------------------------------
extern "C" void kernel_launch(void* const* d_in, const int* in_sizes, int n_in,
                              void* d_out, int out_size, void* d_ws, size_t ws_size,
                              hipStream_t stream) {
    const float* index  = (const float*)d_in[0];
    const float* memory = (const float*)d_in[1];
    const float* Wq     = (const float*)d_in[2];
    const float* Wk     = (const float*)d_in[3];
    const float* Wv     = (const float*)d_in[4];

    char* ws = (char*)d_ws;
    const size_t wtElems  = (size_t)3 * HEAD * N_EMBED;          // 196608
    const size_t qkvElems = (size_t)BATCH * TLEN * HEAD;         // 1048576
    _Float16* WT   = (_Float16*)ws;
    _Float16* q16  = (_Float16*)(ws + wtElems * 2);
    _Float16* k16  = (_Float16*)(ws + (wtElems + qkvElems) * 2);
    _Float16* vT16 = (_Float16*)(ws + (wtElems + 2 * qkvElems) * 2);

    const _Float16* WqT = WT;
    const _Float16* WkT = WT + (size_t)HEAD * N_EMBED;
    const _Float16* WvT = WT + (size_t)2 * HEAD * N_EMBED;

    const int tilesPerB = TLEN / 16;

    wtrans_kernel<<<(3 * HEAD * N_EMBED + 255) / 256, 256, 0, stream>>>(Wq, Wk, Wv, WT);
    proj_q_kernel<<<BATCH * tilesPerB, 32, 0, stream>>>(index, WqT, q16);
    proj_kv_kernel<<<BATCH * tilesPerB, 32, 0, stream>>>(memory, WkT, WvT, k16, vT16);
    attn_kernel<<<BATCH * tilesPerB, 32, 0, stream>>>(q16, k16, vT16, (float*)d_out);
}
